// FeatureMixerHead_71691594105028
// MI455X (gfx1250) — matrix operand order repaired and checked by reading, not compiled, not run on hardware
//
#include <hip/hip_runtime.h>
#include <hip/hip_bf16.h>
#include <math.h>

typedef __attribute__((ext_vector_type(16))) __bf16 v16bf;
typedef __attribute__((ext_vector_type(8)))  float  v8f;

#define NQT   1800          // B*NQ
#define MQ    1808          // padded to 16
#define QD    256
#define NH    4
#define NP    32
#define CD    64
#define ENP   64
#define P1C   16384         // FEATS_DIM*ENP
#define P2C   8192          // ENP*NP*NH
#define PTOT  24576
#define KOUT  16384
#define LNEPS 1e-5f

// ---- gfx1250 async memory->LDS path (guarded; falls back to manual copy) ----
#if defined(__has_builtin)
#  if __has_builtin(__builtin_amdgcn_global_load_async_to_lds_b128)
#    define HAVE_ASYNC_LDS 1
#  endif
#endif
#ifndef HAVE_ASYNC_LDS
#  define HAVE_ASYNC_LDS 0
#endif

typedef __attribute__((address_space(1))) char  gchar;
typedef __attribute__((address_space(3))) char  lchar;
// builtin expects pointers to 'int __vector(4)' (per hipcc diagnostic)
typedef int v4i_vs __attribute__((vector_size(16)));
typedef __attribute__((address_space(1))) v4i_vs gv4;
typedef __attribute__((address_space(3))) v4i_vs lv4;

__device__ inline void wait_async_lds() {
#if HAVE_ASYNC_LDS
#  if __has_builtin(__builtin_amdgcn_s_wait_asynccnt)
  __builtin_amdgcn_s_wait_asynccnt(0);
#  else
  asm volatile("s_wait_asynccnt 0" ::: "memory");
#  endif
#endif
}

// ---------------- fragment helpers (16-bit WMMA operands, contiguous-K order used here) ----------------
// A 16x32 bf16: lane m=l&15, hi=l>>4. elems 0..7 -> K = kb+hi*16+j ; 8..15 -> K = kb+hi*16+8+j
__device__ inline v16bf ldA(const __bf16* p0, const __bf16* p1) {
  v16bf r;
#pragma unroll
  for (int j = 0; j < 8; ++j) { r[j] = p0[j]; r[j + 8] = p1[j]; }
  return r;
}
// B 32x16 bf16: lane n=l&15, hi=l>>4. elem j -> K = kb + hi*16 + j (16 contiguous K at fixed N)
__device__ inline v16bf ldB(const __bf16* p) {
  v16bf r;
#pragma unroll
  for (int j = 0; j < 16; ++j) r[j] = p[j];
  return r;
}
#define WMMA_BF16(a, b, c) \
  __builtin_amdgcn_wmma_f32_16x16x32_bf16(false, (a), false, (b), (short)0, (c), false, false)

__device__ inline float gelu_exact(float x) {
  return 0.5f * x * (1.0f + erff(x * 0.70710678118654752f));
}
__device__ inline float wave_reduce_sum(float v) {
#pragma unroll
  for (int o = 16; o > 0; o >>= 1) v += __shfl_xor(v, o, 32);
  return v;
}

// ---------------- prep kernels ----------------
__global__ void cvt_pad_qbf(const float* __restrict__ q, __bf16* __restrict__ qbf) {
  int idx = blockIdx.x * 256 + threadIdx.x;
  if (idx >= MQ * QD) return;
  int row = idx >> 8;
  qbf[idx] = (row < NQT) ? (__bf16)q[idx] : (__bf16)0.0f;
}

// src f32 [K][N] row-major -> dst bf16 [N][K]
__global__ void transpose_cvt(const float* __restrict__ src, __bf16* __restrict__ dst,
                              int K, int N) {
  size_t idx = (size_t)blockIdx.x * 256 + threadIdx.x;
  if (idx >= (size_t)K * N) return;
  int n = (int)(idx % N), k = (int)(idx / N);
  dst[(size_t)n * K + k] = (__bf16)src[idx];
}

__global__ void zero_f32(float* __restrict__ p, int n) {
  int i = blockIdx.x * 256 + threadIdx.x;
  if (i < n) p[i] = 0.0f;
}

// ---------------- offsets GEMM (f32 VALU: coords are precision-sensitive) ----------------
__global__ void offs_kernel(const float* __restrict__ query, const float* __restrict__ W_off,
                            const float* __restrict__ b_off, float* __restrict__ offs) {
  __shared__ float qrow[QD];
  int q = blockIdx.x, n = threadIdx.x;
  qrow[n] = query[(size_t)q * QD + n];
  __syncthreads();
  float s = b_off[n];
#pragma unroll 8
  for (int k = 0; k < QD; ++k) s = fmaf(qrow[k], W_off[(size_t)k * QD + n], s);
  offs[(size_t)q * QD + n] = s;
}

// ---------------- deformable sampling: block=(q), wave=head, lane=point ----------------
__global__ void __launch_bounds__(128)
sample_kernel(const float* __restrict__ f0, const float* __restrict__ f1,
              const float* __restrict__ f2, const float* __restrict__ f3,
              const float* __restrict__ bboxes, const float* __restrict__ offs,
              const float* __restrict__ W_lvl, const float* __restrict__ b_lvl,
              float* __restrict__ S) {
  int q = blockIdx.x;
  int h = threadIdx.x >> 5, p = threadIdx.x & 31;
  int b = q / 900;
  const float* bb = bboxes + (size_t)q * 4;
  float cx = bb[0], cy = bb[1];
  float sc = exp2f(bb[2]);
  float rw = sc * exp2f(-0.5f * bb[3]);
  float rh = sc * exp2f(0.5f * bb[3]);
  // level softmax (depends only on scale -> same for all h,p of this query)
  float lg[4], lw[4];
#pragma unroll
  for (int i = 0; i < 4; ++i) lg[i] = sc * W_lvl[i] + b_lvl[i];
  float mx = fmaxf(fmaxf(lg[0], lg[1]), fmaxf(lg[2], lg[3]));
  float den = 0.f;
#pragma unroll
  for (int i = 0; i < 4; ++i) { lw[i] = expf(lg[i] - mx); den += lw[i]; }
  float rden = 1.0f / den;
#pragma unroll
  for (int i = 0; i < 4; ++i) lw[i] *= rden;

  int hp = h * NP + p;
  float ox = offs[(size_t)q * QD + hp * 2 + 0];
  float oy = offs[(size_t)q * QD + hp * 2 + 1];
  float sx = cx + ox * rw, sy = cy + oy * rh;

  float acc[CD];
#pragma unroll
  for (int c = 0; c < CD; ++c) acc[c] = 0.0f;

  const float* fptr[4] = {f0, f1, f2, f3};
  const int   Hs[4]    = {200, 100, 50, 25};
  const float invs[4]  = {0.25f, 0.125f, 0.0625f, 0.03125f};

#pragma unroll
  for (int l = 0; l < 4; ++l) {
    int H = Hs[l], HW = H * H;
    float xf = sx * invs[l] - 0.5f, yf = sy * invs[l] - 0.5f;
    float x0f = floorf(xf), y0f = floorf(yf);
    int x0 = (int)x0f, y0 = (int)y0f, x1 = x0 + 1, y1 = y0 + 1;
    float wx1 = xf - x0f, wx0 = 1.f - wx1, wy1 = yf - y0f, wy0 = 1.f - wy1;
    float L = lw[l];
    float v00 = (x0 >= 0 && x0 < H && y0 >= 0 && y0 < H) ? 1.f : 0.f;
    float v10 = (x1 >= 0 && x1 < H && y0 >= 0 && y0 < H) ? 1.f : 0.f;
    float v01 = (x0 >= 0 && x0 < H && y1 >= 0 && y1 < H) ? 1.f : 0.f;
    float v11 = (x1 >= 0 && x1 < H && y1 >= 0 && y1 < H) ? 1.f : 0.f;
    float w00 = wx0 * wy0 * L * v00, w10 = wx1 * wy0 * L * v10;
    float w01 = wx0 * wy1 * L * v01, w11 = wx1 * wy1 * L * v11;
    int cx0 = min(max(x0, 0), H - 1), cx1 = min(max(x1, 0), H - 1);
    int cy0 = min(max(y0, 0), H - 1), cy1 = min(max(y1, 0), H - 1);
    int o00 = cy0 * H + cx0, o10 = cy0 * H + cx1;
    int o01 = cy1 * H + cx0, o11 = cy1 * H + cx1;
    const float* base = fptr[l] + (size_t)(b * 256 + h * CD) * HW;
#pragma unroll
    for (int c = 0; c < CD; ++c) {
      const float* pc = base + (size_t)c * HW;
      float a = acc[c];
      a = fmaf(w00, pc[o00], a);
      a = fmaf(w10, pc[o10], a);
      a = fmaf(w01, pc[o01], a);
      a = fmaf(w11, pc[o11], a);
      acc[c] = a;
    }
  }
  float* dst = S + (((size_t)(q * NH + h)) * NP + p) * CD;
#pragma unroll
  for (int c = 0; c < CD; ++c) dst[c] = acc[c];
}

// ---------------- generic bf16 WMMA GEMM: C[M x N] = A[M x K] * Bt[N][K] ----------------
// 1 wave computes a 16x64 slab (4 n-tiles). KSPLIT>1 -> atomicAdd f32 partials.
template <bool OUT_BF16, int KSPLIT>
__global__ void __launch_bounds__(256)
gemm_nt4(const __bf16* __restrict__ A, const __bf16* __restrict__ Bt,
         const float* __restrict__ bias, void* __restrict__ Cout,
         int Mtiles, int Ntg, int K, int N) {
  int job = blockIdx.x * (blockDim.x >> 5) + (threadIdx.x >> 5);
  if (job >= Mtiles * Ntg * KSPLIT) return;
  int ks = job % KSPLIT;
  int t = job / KSPLIT;
  int mt = t / Ntg, ntg = t % Ntg;
  int lane = threadIdx.x & 31, m16 = lane & 15, hi = lane >> 4;
  int kchunk = K / KSPLIT, k0 = ks * kchunk;

  const __bf16* arow = A + (size_t)(mt * 16 + m16) * K + k0;
  const __bf16* brow[4];
#pragma unroll
  for (int j = 0; j < 4; ++j)
    brow[j] = Bt + (size_t)(ntg * 64 + j * 16 + m16) * K + k0;

  v8f acc[4] = {};
#pragma unroll 2
  for (int kb = 0; kb < kchunk; kb += 32) {
    // speculative prefetch of next K-chunk (silently dropped past the end)
    __builtin_prefetch((const void*)(arow + kb + 32), 0, 1);
#pragma unroll
    for (int j = 0; j < 4; ++j)
      __builtin_prefetch((const void*)(brow[j] + kb + 32), 0, 1);
    v16bf a = ldA(arow + kb + hi * 16, arow + kb + hi * 16 + 8);
#pragma unroll
    for (int j = 0; j < 4; ++j) {
      v16bf bb = ldB(brow[j] + kb + hi * 16);
      acc[j] = WMMA_BF16(a, bb, acc[j]);
    }
  }
#pragma unroll
  for (int j = 0; j < 4; ++j) {
#pragma unroll
    for (int r = 0; r < 8; ++r) {
      int row = mt * 16 + hi * 8 + r;
      int col = ntg * 64 + j * 16 + m16;
      size_t idx = (size_t)row * N + col;
      if constexpr (KSPLIT == 1) {
        float v = acc[j][r] + (bias ? bias[col] : 0.0f);
        if constexpr (OUT_BF16) ((__bf16*)Cout)[idx] = (__bf16)v;
        else                    ((float*)Cout)[idx] = v;
      } else {
        atomicAdd(((float*)Cout) + idx, acc[j][r]);
      }
    }
  }
}

// ---------------- dynamic mixer: block = (q,h), 4 waves ----------------
__global__ void __launch_bounds__(128)
mixer_kernel(const float* __restrict__ S, const __bf16* __restrict__ params,
             __bf16* __restrict__ M2) {
  __shared__ __bf16 sA[32 * 64];    // sampled feats [p][c]
  __shared__ __bf16 p1t[64 * 64];   // p1 transposed [e][c]  (B-frag contiguous K)
  __shared__ __bf16 p2s[64 * 32];   // p2 [e][p]             (A-frag contiguous K)
  __shared__ __bf16 o1t[64 * 32];   // stage-1 out transposed [f][p]
  __shared__ float red[8], red2[8];
  int q = blockIdx.x >> 2, h = blockIdx.x & 3;
  int tid = threadIdx.x, wid = tid >> 5, lane = tid & 31, m16 = lane & 15, hi = lane >> 4;

  const float*  Sb  = S + (size_t)(q * NH + h) * (NP * CD);
  const __bf16* pr  = params + (size_t)q * PTOT + (size_t)h * (CD * ENP);
  const __bf16* p2g = params + (size_t)q * PTOT + P1C + (size_t)h * (ENP * NP);

#if HAVE_ASYNC_LDS
  {
    // p2 block: 4096 B contiguous -> async DMA straight into LDS (no VGPR bounce)
    gchar* gsrc = (gchar*)(void*)(const void*)p2g;
    lchar* ldst = (lchar*)(void*)p2s;
    for (int i = tid; i < 256; i += 128)
      __builtin_amdgcn_global_load_async_to_lds_b128(
          (gv4*)(gsrc + i * 16), (lv4*)(ldst + i * 16), 0, 0);
  }
#else
  for (int i = tid; i < 2048; i += 128) p2s[i] = p2g[i];
#endif
  for (int i = tid; i < 2048; i += 128) sA[i] = (__bf16)Sb[i];
  for (int i = tid; i < 4096; i += 128) { int c = i >> 6, e = i & 63; p1t[e * 64 + c] = pr[i]; }
  wait_async_lds();
  __syncthreads();

  // ---- stage 1: o1[32p x 64e] = S[32p x 64c] @ p1[64c x 64e] ----
  int mt = wid & 1;
  bool act = wid < 2;
  v8f acc[4] = {};
  const __bf16* arow = sA + (mt * 16 + m16) * 64;
#pragma unroll
  for (int kb = 0; kb < 64; kb += 32) {
    v16bf a = ldA(arow + kb + hi * 16, arow + kb + hi * 16 + 8);
#pragma unroll
    for (int nt = 0; nt < 4; ++nt) {
      v16bf bb = ldB(p1t + (nt * 16 + m16) * 64 + kb + hi * 16);
      acc[nt] = WMMA_BF16(a, bb, acc[nt]);
    }
  }
  float sum = 0.f, ss = 0.f;
#pragma unroll
  for (int nt = 0; nt < 4; ++nt)
#pragma unroll
    for (int r = 0; r < 8; ++r) {
      float v = gelu_exact(acc[nt][r]);
      acc[nt][r] = v; sum += v; ss += v * v;
    }
  if (!act) { sum = 0.f; ss = 0.f; }
  sum = wave_reduce_sum(sum); ss = wave_reduce_sum(ss);
  if (lane == 0) { red[wid] = sum; red[4 + wid] = ss; }
  __syncthreads();
  {
    float tot = red[0] + red[1] + red[2] + red[3];
    float tss = red[4] + red[5] + red[6] + red[7];
    float mean = tot * (1.f / 2048.f);
    float inv = rsqrtf(tss * (1.f / 2048.f) - mean * mean + LNEPS);
    if (act) {
#pragma unroll
      for (int nt = 0; nt < 4; ++nt)
#pragma unroll
        for (int r = 0; r < 8; ++r) {
          int pp = mt * 16 + hi * 8 + r;    // row (point)
          int f  = nt * 16 + m16;           // col
          o1t[f * 32 + pp] = (__bf16)((acc[nt][r] - mean) * inv);
        }
    }
  }
  __syncthreads();

  // ---- stage 2: o2[64e x 64f] = p2[64e x 32p] @ o1n[32p x 64f] ----
  v8f acc2[4] = {};
  const __bf16* a2 = p2s + (wid * 16 + m16) * 32;
  v16bf a = ldA(a2 + hi * 16, a2 + hi * 16 + 8);
#pragma unroll
  for (int nt = 0; nt < 4; ++nt) {
    v16bf bb = ldB(o1t + (nt * 16 + m16) * 32 + hi * 16);
    acc2[nt] = WMMA_BF16(a, bb, acc2[nt]);
  }
  sum = 0.f; ss = 0.f;
#pragma unroll
  for (int nt = 0; nt < 4; ++nt)
#pragma unroll
    for (int r = 0; r < 8; ++r) {
      float v = gelu_exact(acc2[nt][r]);
      acc2[nt][r] = v; sum += v; ss += v * v;
    }
  sum = wave_reduce_sum(sum); ss = wave_reduce_sum(ss);
  if (lane == 0) { red2[wid] = sum; red2[4 + wid] = ss; }
  __syncthreads();
  {
    float tot = red2[0] + red2[1] + red2[2] + red2[3];
    float tss = red2[4] + red2[5] + red2[6] + red2[7];
    float mean = tot * (1.f / 4096.f);
    float inv = rsqrtf(tss * (1.f / 4096.f) - mean * mean + LNEPS);
    __bf16* dst = M2 + (size_t)q * KOUT + (size_t)h * 4096;
#pragma unroll
    for (int nt = 0; nt < 4; ++nt)
#pragma unroll
      for (int r = 0; r < 8; ++r) {
        int e = wid * 16 + hi * 8 + r;
        int f = nt * 16 + m16;
        dst[e * 64 + f] = (__bf16)((acc2[nt][r] - mean) * inv);
      }
  }
}

// ---------------- epilogue: +b_out, LN(g_mix), residual, LN(g_head) ----------------
__global__ void __launch_bounds__(256)
epilogue_kernel(const float* __restrict__ gout, const float* __restrict__ b_out,
                const float* __restrict__ query, const float* __restrict__ g_mix,
                const float* __restrict__ bt_mix, const float* __restrict__ g_head,
                const float* __restrict__ bt_head, float* __restrict__ out) {
  int wid = threadIdx.x >> 5, lane = threadIdx.x & 31;
  int q = blockIdx.x * 8 + wid;
  if (q >= NQT) return;
  const float* row = gout + (size_t)q * QD;
  float t[8], s = 0.f, ss = 0.f;
#pragma unroll
  for (int j = 0; j < 8; ++j) {
    int n = lane + j * 32;
    t[j] = row[n] + b_out[n];
    s += t[j]; ss += t[j] * t[j];
  }
  s = wave_reduce_sum(s); ss = wave_reduce_sum(ss);
  float mean = s * (1.f / 256.f);
  float inv = rsqrtf(ss * (1.f / 256.f) - mean * mean + LNEPS);
  float s2 = 0.f, ss2 = 0.f;
#pragma unroll
  for (int j = 0; j < 8; ++j) {
    int n = lane + j * 32;
    float v = (t[j] - mean) * inv * g_mix[n] + bt_mix[n];
    v += query[(size_t)q * QD + n];
    t[j] = v; s2 += v; ss2 += v * v;
  }
  s2 = wave_reduce_sum(s2); ss2 = wave_reduce_sum(ss2);
  float mean2 = s2 * (1.f / 256.f);
  float inv2 = rsqrtf(ss2 * (1.f / 256.f) - mean2 * mean2 + LNEPS);
#pragma unroll
  for (int j = 0; j < 8; ++j) {
    int n = lane + j * 32;
    out[(size_t)q * QD + n] = (t[j] - mean2) * inv2 * g_head[n] + bt_head[n];
  }
}

extern "C" void kernel_launch(void* const* d_in, const int* in_sizes, int n_in,
                              void* d_out, int out_size, void* d_ws, size_t ws_size,
                              hipStream_t stream) {
  const float* feat0  = (const float*)d_in[0];
  const float* feat1  = (const float*)d_in[1];
  const float* feat2  = (const float*)d_in[2];
  const float* feat3  = (const float*)d_in[3];
  const float* query  = (const float*)d_in[4];
  const float* bboxes = (const float*)d_in[5];
  const float* W_off  = (const float*)d_in[6];
  const float* b_off  = (const float*)d_in[7];
  const float* W_lvl  = (const float*)d_in[8];
  const float* b_lvl  = (const float*)d_in[9];
  const float* W_pg   = (const float*)d_in[10];
  const float* b_pg   = (const float*)d_in[11];
  const float* W_out  = (const float*)d_in[12];
  const float* b_out  = (const float*)d_in[13];
  const float* g_mix  = (const float*)d_in[14];
  const float* bt_mix = (const float*)d_in[15];
  const float* g_head = (const float*)d_in[16];
  const float* bt_head= (const float*)d_in[17];
  float* out = (float*)d_out;

  char* ws = (char*)d_ws;
  size_t off = 0;
  auto take = [&](size_t bytes) -> char* {
    char* p = ws + off;
    off = (off + bytes + 255) & ~(size_t)255;
    return p;
  };
  __bf16* qbf    = (__bf16*)take((size_t)MQ * QD * 2);
  float*  offs   = (float*) take((size_t)NQT * QD * 4);
  float*  S      = (float*) take((size_t)NQT * NH * NP * CD * 4);
  __bf16* wpgt   = (__bf16*)take((size_t)PTOT * QD * 2);
  __bf16* params = (__bf16*)take((size_t)MQ * PTOT * 2);
  __bf16* M2     = (__bf16*)take((size_t)MQ * KOUT * 2);
  __bf16* woutt  = (__bf16*)take((size_t)QD * KOUT * 2);
  float*  gout   = (float*) take((size_t)MQ * QD * 4);

  // prep
  cvt_pad_qbf<<<(MQ * QD + 255) / 256, 256, 0, stream>>>(query, qbf);
  transpose_cvt<<<(int)(((size_t)QD * PTOT + 255) / 256), 256, 0, stream>>>(W_pg, wpgt, QD, PTOT);
  transpose_cvt<<<(int)(((size_t)KOUT * QD + 255) / 256), 256, 0, stream>>>(W_out, woutt, KOUT, QD);

  // offsets (f32) + sampling
  offs_kernel<<<NQT, QD, 0, stream>>>(query, W_off, b_off, offs);
  sample_kernel<<<NQT, 128, 0, stream>>>(feat0, feat1, feat2, feat3, bboxes, offs,
                                         W_lvl, b_lvl, S);

  // params GEMM: [MQ x 256] @ [256 x 24576] -> bf16
  {
    int Mtiles = MQ / 16, Ntg = PTOT / 64;
    int jobs = Mtiles * Ntg;
    gemm_nt4<true, 1><<<(jobs + 7) / 8, 256, 0, stream>>>(qbf, wpgt, b_pg, params,
                                                          Mtiles, Ntg, QD, PTOT);
  }

  // per-query dynamic mixing
  mixer_kernel<<<NQT * NH, 128, 0, stream>>>(S, params, M2);

  // final GEMM: [MQ x 16384] @ [16384 x 256] with K-split + atomic accumulate
  zero_f32<<<(MQ * QD + 255) / 256, 256, 0, stream>>>(gout, MQ * QD);
  {
    int Mtiles = MQ / 16, Ntg = QD / 64;
    int jobs = Mtiles * Ntg * 32;
    gemm_nt4<false, 32><<<(jobs + 7) / 8, 256, 0, stream>>>(M2, woutt, nullptr, gout,
                                                            Mtiles, Ntg, KOUT, QD);
  }

  // epilogue LNs + residual
  epilogue_kernel<<<(NQT + 7) / 8, 256, 0, stream>>>(gout, b_out, query, g_mix, bt_mix,
                                                     g_head, bt_head, out);
}
